// QuantumOlfactoryReceptor_36567351558870
// MI455X (gfx1250) — matrix operand order repaired and checked by reading, not compiled, not run on hardware
//
#include <hip/hip_runtime.h>

// ---------------------------------------------------------------------------
// Types for CDNA5 WMMA (wave32)
// ---------------------------------------------------------------------------
typedef __attribute__((ext_vector_type(16))) __bf16 v16bf;
typedef __attribute__((ext_vector_type(8)))  float  v8f;

union FragBF16 {
    v16bf v;
    unsigned short s[16];
};

__device__ __forceinline__ unsigned short f2bf(float f) {
    union { float f; unsigned u; } x;
    x.f = f;
    unsigned u = x.u + 0x7FFFu + ((x.u >> 16) & 1u);  // round-to-nearest-even
    return (unsigned short)(u >> 16);
}

// fast sigmoid / tanh on v_exp_f32 + v_rcp_f32 (no IEEE division ladder)
__device__ __forceinline__ float fast_sigmoid(float x) {
    return __builtin_amdgcn_rcpf(1.0f + __expf(-x));
}
__device__ __forceinline__ float fast_tanh(float x) {
    float ax = fabsf(x);
    ax = fminf(ax, 15.0f);
    float t = __expf(-2.0f * ax);
    float r = (1.0f - t) * __builtin_amdgcn_rcpf(1.0f + t);
    return copysignf(r, x);
}

// ---------------------------------------------------------------------------
// Generic GEMM:  out[M,N] = act( A[M,K] * W[N,K]^T + bias1 + bias2 (+ out) )
// A fp32 (amode==1: synthesized LSTM layer-0 input: receptor*gsig*adapt+noise)
// fp32 -> bf16 in LDS, consumed by v_wmma_f32_16x16x32_bf16.
// Block: 256 threads = 8 waves; block tile M=64 x N=256; K step 32.
// Waves arranged 2(M) x 4(N); wave tile 32x64 = 2x4 WMMA subtiles.
// ---------------------------------------------------------------------------
__global__ __launch_bounds__(256)
void gemm_bf16_wmma(const float* __restrict__ A, const float* __restrict__ W,
                    const float* __restrict__ bias1, const float* __restrict__ bias2,
                    float* __restrict__ Cout,
                    int M, int N, int K, int accumulate, int act,
                    int amode, const float* __restrict__ noisep,
                    const float* __restrict__ gsig, int tstep, float adapt) {
    const int tid  = threadIdx.x;
    const int lane = tid & 31;
    const int wave = tid >> 5;
    const int waveM = (wave >> 2) * 32;   // 0 or 32
    const int waveN = (wave & 3) * 64;    // 0,64,128,192
    const int blockM = blockIdx.y * 64;
    const int blockN = blockIdx.x * 256;

    __shared__ unsigned short As[64][34];
    __shared__ unsigned short Bs[256][34];

    v8f zero8 = {0.f, 0.f, 0.f, 0.f, 0.f, 0.f, 0.f, 0.f};
    v8f acc[2][4];
#pragma unroll
    for (int i = 0; i < 2; ++i)
#pragma unroll
        for (int j = 0; j < 4; ++j) acc[i][j] = zero8;

    for (int k0 = 0; k0 < K; k0 += 32) {
        // ---- load A tile 64x32 (8 elems/thread) -> bf16 LDS ----
        {
            int row = tid >> 2;
            int kch = (tid & 3) * 8;
            int gr  = blockM + row;
#pragma unroll
            for (int j = 0; j < 8; ++j) {
                int k = k0 + kch + j;
                float v = 0.f;
                if (k < K && gr < M) {
                    if (amode == 0) {
                        v = A[(size_t)gr * K + k];
                    } else {
                        // layer-0 LSTM input: receptor*sig(gpc)*adapt + noise[b,t,r]
                        float rec = A[(size_t)gr * 400 + k];
                        v = rec * gsig[k] * adapt +
                            noisep[((size_t)gr * 50 + tstep) * 400 + k];
                    }
                }
                As[row][kch + j] = f2bf(v);
            }
        }
        // ---- load B tile 256x32 (W rows along N; 32 elems/thread) ----
        {
            int n  = tid;           // 0..255
            int gn = blockN + n;
            if (gn < N && k0 + 32 < K)
                __builtin_prefetch(&W[(size_t)gn * K + k0 + 32], 0, 1);
#pragma unroll
            for (int j = 0; j < 32; ++j) {
                int k = k0 + j;
                float v = (k < K && gn < N) ? W[(size_t)gn * K + k] : 0.f;
                Bs[n][j] = f2bf(v);
            }
        }
        __syncthreads();

        // ---- per-lane fragments (ISA 16-bit A/B layouts) ----
        const int r  = lane & 15;
        const int hf = lane >> 4;
        v16bf afrag[2], bfrag[4];
#pragma unroll
        for (int ms = 0; ms < 2; ++ms) {
            FragBF16 u;
            const unsigned short* ap = &As[waveM + ms * 16 + r][hf * 8];
#pragma unroll
            for (int j = 0; j < 8; ++j) { u.s[j] = ap[j]; u.s[8 + j] = ap[16 + j]; }
            afrag[ms] = u.v;
        }
#pragma unroll
        for (int ns = 0; ns < 4; ++ns) {
            FragBF16 u;
            const unsigned short* bp = &Bs[waveN + ns * 16 + r][hf * 8];
#pragma unroll
            for (int j = 0; j < 8; ++j) { u.s[j] = bp[j]; u.s[8 + j] = bp[16 + j]; }
            bfrag[ns] = u.v;
        }
#pragma unroll
        for (int ms = 0; ms < 2; ++ms)
#pragma unroll
            for (int ns = 0; ns < 4; ++ns)
                acc[ms][ns] = __builtin_amdgcn_wmma_f32_16x16x32_bf16(
                    false, afrag[ms], false, bfrag[ns], (short)0, acc[ms][ns],
                    false, false);
        __syncthreads();
    }

    // ---- epilogue: bias + optional accumulate + activation ----
    const int nl = lane & 15;
    const int mh = (lane >> 4) * 8;
#pragma unroll
    for (int ms = 0; ms < 2; ++ms) {
#pragma unroll
        for (int ns = 0; ns < 4; ++ns) {
            int n = blockN + waveN + ns * 16 + nl;
            if (n < N) {
                float bv = (bias1 ? bias1[n] : 0.f) + (bias2 ? bias2[n] : 0.f);
#pragma unroll
                for (int rr = 0; rr < 8; ++rr) {
                    int m = blockM + waveM + ms * 16 + mh + rr;
                    if (m < M) {
                        float v = acc[ms][ns][rr] + bv;
                        if (accumulate) v += Cout[(size_t)m * N + n];
                        if (act == 1)      v = fmaxf(v, 0.f);
                        else if (act == 2) v = fast_tanh(v);
                        else if (act == 3) v = fast_sigmoid(v);
                        Cout[(size_t)m * N + n] = v;
                    }
                }
            }
        }
    }
}

// ---------------------------------------------------------------------------
// Quantum evolution: 16 lanes per sample (2 samples/wave). Lane i holds row i
// of H and amplitude element i.  100 steps of order-12 Taylor action of
// expm(-i*Ht*DT) on v, with exp(-t/COH) decay.
// ---------------------------------------------------------------------------
__global__ __launch_bounds__(256)
void quantum_kernel(const float* __restrict__ amp_re, const float* __restrict__ amp_im,
                    const float* __restrict__ Hre, const float* __restrict__ Him,
                    const float* __restrict__ mol,
                    float* __restrict__ obs, float* __restrict__ ampf) {
    const int gtid = blockIdx.x * blockDim.x + threadIdx.x;
    const int lane = threadIdx.x & 31;
    const int row  = lane & 15;
    const int samp = gtid >> 4;   // grid is exactly B*16 threads

    float hre[16], him[16];
#pragma unroll
    for (int j = 0; j < 16; ++j) {
        hre[j] = Hre[row * 16 + j];
        him[j] = Him[row * 16 + j];
    }
    const float mol_i = mol[(size_t)samp * 512 + row];

    float vr = amp_re[samp * 16 + row];
    float vi = amp_im[samp * 16 + row];

    // normalize over the 16-lane half
    float s = vr * vr + vi * vi;
#pragma unroll
    for (int o = 8; o >= 1; o >>= 1) s += __shfl_xor(s, o, 16);
    float inv = rsqrtf(s);
    vr *= inv; vi *= inv;

    const float DT = 0.01f;
    for (int step = 0; step < 100; ++step) {
        float t   = step * DT;
        // A = -i*Ht*DT :  A_re = DT*Him ;  A_im = -DT*Hre*(1 + mol_i*t)
        float fac = -DT * (1.0f + mol_i * t);
        float yr = vr, yi = vi, wr = vr, wi = vi;
        for (int k = 1; k <= 12; ++k) {
            float nr = 0.f, ni = 0.f;
#pragma unroll
            for (int j = 0; j < 16; ++j) {
                float br = __shfl(wr, (lane & 16) | j, 32);
                float bi = __shfl(wi, (lane & 16) | j, 32);
                float ar = DT * him[j];
                float ai = fac * hre[j];
                nr += ar * br - ai * bi;
                ni += ar * bi + ai * br;
            }
            float invk = __builtin_amdgcn_rcpf((float)k);
            wr = nr * invk; wi = ni * invk;
            yr += wr; yi += wi;
        }
        float dec = __expf(-t * 0.02f);   // exp(-t/50)
        vr = yr * dec; vi = yi * dec;
    }
    obs[samp * 16 + row]        = vr * vr + vi * vi;
    ampf[samp * 32 + row]       = vr;
    ampf[samp * 32 + 16 + row]  = vi;
}

// ---------------------------------------------------------------------------
// LSTM gate update (PyTorch order i,f,g,o).  Z:[B,1024], h/c:[B,256]
// ---------------------------------------------------------------------------
__global__ void lstm_gate(const float* __restrict__ Z, float* __restrict__ c,
                          float* __restrict__ h, float* __restrict__ seq_out,
                          int t, int useC) {
    int idx = blockIdx.x * blockDim.x + threadIdx.x;  // B*256
    int b = idx >> 8, hh = idx & 255;
    const float* z = Z + (size_t)b * 1024;
    float ig = fast_sigmoid(z[hh]);
    float fg = fast_sigmoid(z[256 + hh]);
    float gg = fast_tanh(z[512 + hh]);
    float og = fast_sigmoid(z[768 + hh]);
    float cv = useC ? c[idx] : 0.0f;
    cv = fg * cv + ig * gg;
    float hv = og * fast_tanh(cv);
    c[idx] = cv;
    h[idx] = hv;
    if (seq_out) seq_out[((size_t)b * 50 + t) * 256 + hh] = hv;
}

// sigmoid over a small vector (precomputes sig(gpc))
__global__ void sigmoid_vec(const float* __restrict__ in, float* __restrict__ out, int n) {
    int i = blockIdx.x * blockDim.x + threadIdx.x;
    if (i < n) out[i] = fast_sigmoid(in[i]);
}

// concat [obs(16) | mol(512)] -> bio[B,528]
__global__ void concat_bio(const float* __restrict__ obs, const float* __restrict__ mol,
                           float* __restrict__ bio) {
    int idx = blockIdx.x * blockDim.x + threadIdx.x;  // B*528
    if (idx >= 2048 * 528) return;
    int b = idx / 528, j = idx % 528;
    bio[idx] = (j < 16) ? obs[b * 16 + j] : mol[(size_t)b * 512 + (j - 16)];
}

// concat [hT(256) | receptor(400)] -> cf[B,656]
__global__ void concat_cf(const float* __restrict__ hT, const float* __restrict__ rec,
                          float* __restrict__ cf) {
    int idx = blockIdx.x * blockDim.x + threadIdx.x;  // B*656
    if (idx >= 2048 * 656) return;
    int b = idx / 656, j = idx % 656;
    cf[idx] = (j < 256) ? hT[b * 256 + j] : rec[(size_t)b * 400 + (j - 256)];
}

__global__ void zero_scalars(float* a, float* b) {
    if (threadIdx.x == 0 && blockIdx.x == 0) { *a = 0.f; *b = 0.f; }
}

// valence[b] = mean(tanh(seq[b,:,:]))  (12800 elems)
__global__ __launch_bounds__(256)
void valence_kernel(const float* __restrict__ seq, float* __restrict__ val) {
    int b = blockIdx.x;
    const float* p = seq + (size_t)b * 12800;
    float s = 0.f;
    for (int i = threadIdx.x; i < 12800; i += 256) s += fast_tanh(p[i]);
    __shared__ float red[256];
    red[threadIdx.x] = s;
    __syncthreads();
    for (int o = 128; o > 0; o >>= 1) {
        if (threadIdx.x < o) red[threadIdx.x] += red[threadIdx.x + o];
        __syncthreads();
    }
    if (threadIdx.x == 0) val[b] = red[0] / 12800.f;
}

// cons = mean(sigmoid(c3 @ cw4^T + cb4))
__global__ __launch_bounds__(256)
void cons_kernel(const float* __restrict__ c3, const float* __restrict__ w,
                 const float* __restrict__ bias, float* __restrict__ out) {
    int b = blockIdx.x * blockDim.x + threadIdx.x;  // 2048
    float acc = bias[0];
    for (int j = 0; j < 64; ++j) acc += c3[(size_t)b * 64 + j] * w[j];
    float s = fast_sigmoid(acc);
    __shared__ float red[256];
    red[threadIdx.x] = s;
    __syncthreads();
    for (int o = 128; o > 0; o >>= 1) {
        if (threadIdx.x < o) red[threadIdx.x] += red[threadIdx.x + o];
        __syncthreads();
    }
    if (threadIdx.x == 0) atomicAdd(out, red[0] / 2048.f);
}

// von Neumann entropy via 8x8 real-symmetric embedding of the 4x4 rho + Jacobi
__global__ void entropy_kernel(const float* __restrict__ ampf, float* __restrict__ ent,
                               int B) {
    int b = blockIdx.x * blockDim.x + threadIdx.x;
    if (b >= B) return;
    float ar[16], ai[16];
    for (int i = 0; i < 16; ++i) {
        ar[i] = ampf[b * 32 + i];
        ai[i] = ampf[b * 32 + 16 + i];
    }
    float S[8][8];
    for (int j = 0; j < 4; ++j)
        for (int k = 0; k < 4; ++k) {
            float re = 0.f, im = 0.f;
            for (int i = 0; i < 4; ++i) {
                float xr = ar[i * 4 + j], xi = ai[i * 4 + j];
                float yr = ar[i * 4 + k], yi = ai[i * 4 + k];
                re += xr * yr + xi * yi;   // rs[i,j]*conj(rs[i,k])
                im += xi * yr - xr * yi;
            }
            S[j][k] = re;          S[j][k + 4] = -im;
            S[j + 4][k] = im;      S[j + 4][k + 4] = re;
        }
    for (int sweep = 0; sweep < 8; ++sweep)
        for (int p = 0; p < 7; ++p)
            for (int q = p + 1; q < 8; ++q) {
                float apq = S[p][q];
                if (fabsf(apq) < 1e-12f) continue;
                float tau = (S[q][q] - S[p][p]) / (2.f * apq);
                float tt  = (tau >= 0.f ? 1.f : -1.f) / (fabsf(tau) + sqrtf(1.f + tau * tau));
                float cc  = rsqrtf(1.f + tt * tt), ss = tt * cc;
                for (int i = 0; i < 8; ++i) {
                    float sip = S[i][p], siq = S[i][q];
                    S[i][p] = cc * sip - ss * siq;
                    S[i][q] = ss * sip + cc * siq;
                }
                for (int i = 0; i < 8; ++i) {
                    float spi = S[p][i], sqi = S[q][i];
                    S[p][i] = cc * spi - ss * sqi;
                    S[q][i] = ss * spi + cc * sqi;
                }
            }
    float e = 0.f;
    for (int i = 0; i < 8; ++i) {
        float l = S[i][i];
        if (l < 1e-12f) l = 1e-12f;
        e -= l * __logf(l);
    }
    e *= 0.5f;  // embedding doubles each eigenvalue
    atomicAdd(ent, e / (float)B);
}

// ---------------------------------------------------------------------------
// Host launcher
// ---------------------------------------------------------------------------
extern "C" void kernel_launch(void* const* d_in, const int* in_sizes, int n_in,
                              void* d_out, int out_size, void* d_ws, size_t ws_size,
                              hipStream_t stream) {
    const float* mol   = (const float*)d_in[0];
    const float* noise = (const float*)d_in[1];
    const float* ampr  = (const float*)d_in[2];
    const float* ampi  = (const float*)d_in[3];
    const float* Hre   = (const float*)d_in[4];
    const float* Him   = (const float*)d_in[5];
    const float* gpc   = (const float*)d_in[6];
    const float* bw1 = (const float*)d_in[7],  *bb1 = (const float*)d_in[8];
    const float* bw2 = (const float*)d_in[9],  *bb2 = (const float*)d_in[10];
    const float* bw3 = (const float*)d_in[11], *bb3 = (const float*)d_in[12];
    const float* Wih[3] = {(const float*)d_in[13], (const float*)d_in[17], (const float*)d_in[21]};
    const float* Whh[3] = {(const float*)d_in[14], (const float*)d_in[18], (const float*)d_in[22]};
    const float* bih[3] = {(const float*)d_in[15], (const float*)d_in[19], (const float*)d_in[23]};
    const float* bhh[3] = {(const float*)d_in[16], (const float*)d_in[20], (const float*)d_in[24]};
    const float* cw1 = (const float*)d_in[25], *cb1 = (const float*)d_in[26];
    const float* cw2 = (const float*)d_in[27], *cb2 = (const float*)d_in[28];
    const float* cw3 = (const float*)d_in[29], *cb3 = (const float*)d_in[30];
    const float* cw4 = (const float*)d_in[31], *cb4 = (const float*)d_in[32];

    const int B = 2048;

    // ---- output layout ----
    float* out       = (float*)d_out;
    float* receptor  = out;                          // [B,400]
    float* seq_out   = out + 819200;                 // [B,50,256]
    float* valence   = out + 27033600;               // [B]
    float* cons      = out + 27035648;               // scalar
    float* ent       = out + 27035649;               // scalar

    // ---- workspace layout (floats) ----
    float* ws  = (float*)d_ws;
    float* Z   = ws;                 // 2048*1024
    float* hbuf[3], *cbuf[3];
    {
        float* p = Z + 2097152;
        for (int l = 0; l < 3; ++l) { hbuf[l] = p; p += 524288; cbuf[l] = p; p += 524288; }
    }
    float* obs  = Z + 2097152 + 6 * 524288;          // 2048*16
    float* ampf = obs + 32768;                       // 2048*32
    float* bio  = ampf + 65536;                      // 2048*528
    float* bh1  = bio + 1081344;                     // 2048*256
    float* bh2  = bh1 + 524288;                      // 2048*128
    float* cf   = bh2 + 262144;                      // 2048*656
    float* cc1  = cf + 1343488;                      // 2048*512
    float* cc2  = cc1 + 1048576;                     // 2048*256
    float* cc3  = cc2 + 524288;                      // 2048*64
    float* gsig = cc3 + 131072;                      // 400 (precomputed sigmoid(gpc))

    auto gemm = [&](const float* A, const float* W, const float* b1, const float* b2,
                    float* O, int M, int N, int K, int accum, int act,
                    int amode = 0, const float* np = nullptr, const float* gp = nullptr,
                    int t = 0, float adapt = 0.f) {
        dim3 grid((N + 255) / 256, (M + 63) / 64);
        gemm_bf16_wmma<<<grid, 256, 0, stream>>>(A, W, b1, b2, O, M, N, K, accum, act,
                                                 amode, np, gp, t, adapt);
    };

    // 0) zero atomically-accumulated scalars; precompute sigmoid(gpc)
    zero_scalars<<<1, 32, 0, stream>>>(cons, ent);
    sigmoid_vec<<<2, 256, 0, stream>>>(gpc, gsig, 400);

    // 1) quantum evolution -> obs, final amplitudes
    quantum_kernel<<<(B * 16) / 256, 256, 0, stream>>>(ampr, ampi, Hre, Him, mol, obs, ampf);

    // 2) biological MLP -> receptor
    concat_bio<<<(B * 528 + 255) / 256, 256, 0, stream>>>(obs, mol, bio);
    gemm(bio, bw1, bb1, nullptr, bh1, B, 256, 528, 0, /*relu*/1);
    gemm(bh1, bw2, bb2, nullptr, bh2, B, 128, 256, 0, /*tanh*/2);
    gemm(bh2, bw3, bb3, nullptr, receptor, B, 400, 128, 0, /*sigmoid*/3);

    // 3) stacked LSTM: all 3 layers per timestep (avoids huge scratch)
    for (int t = 0; t < 50; ++t) {
        float adapt = expf(-(float)t / 10.0f);
        for (int l = 0; l < 3; ++l) {
            if (l == 0) {
                // layer-0 input synthesized on the fly from receptor/gsig/noise
                gemm(receptor, Wih[0], bih[0], bhh[0], Z, B, 1024, 400, 0, 0,
                     /*amode*/1, noise, gsig, t, adapt);
            } else {
                gemm(hbuf[l - 1], Wih[l], bih[l], bhh[l], Z, B, 1024, 256, 0, 0);
            }
            if (t > 0) {
                gemm(hbuf[l], Whh[l], nullptr, nullptr, Z, B, 1024, 256, /*accum*/1, 0);
            }
            lstm_gate<<<(B * 256) / 256, 256, 0, stream>>>(
                Z, cbuf[l], hbuf[l], (l == 2) ? seq_out : nullptr, t, t > 0 ? 1 : 0);
        }
    }

    // 4) classifier head
    concat_cf<<<(B * 656 + 255) / 256, 256, 0, stream>>>(hbuf[2], receptor, cf);
    gemm(cf,  cw1, cb1, nullptr, cc1, B, 512, 656, 0, 1);
    gemm(cc1, cw2, cb2, nullptr, cc2, B, 256, 512, 0, 1);
    gemm(cc2, cw3, cb3, nullptr, cc3, B,  64, 256, 0, 2);
    cons_kernel<<<B / 256, 256, 0, stream>>>(cc3, cw4, cb4, cons);

    // 5) valence
    valence_kernel<<<B, 256, 0, stream>>>(seq_out, valence);

    // 6) entanglement entropy
    entropy_kernel<<<(B + 255) / 256, 256, 0, stream>>>(ampf, ent, B);
}
